// LlamaAttention_58995670778391
// MI455X (gfx1250) — matrix operand order repaired and checked by reading, not compiled, not run on hardware
//
#include <hip/hip_runtime.h>

// ---------------------------------------------------------------------------
// MI455X (gfx1250) Llama attention block.
// fp32 -> bf16 once; all GEMMs + flash attention via v_wmma_f32_16x16x32_bf16
// (wave32, f32 accumulate). GEMM waves compute 32x64 tiles with batched
// fragment loads; weight transposes are LDS-tiled for coalescing.
// ---------------------------------------------------------------------------

typedef __bf16 bf16;
typedef __attribute__((ext_vector_type(16))) __bf16 v16bf;
typedef __attribute__((ext_vector_type(8)))  __bf16 v8bf;
typedef __attribute__((ext_vector_type(8)))  float  v8f;

#define BQ   2
#define SQ   2048
#define HQ   4096
#define NHQ  32
#define NKVQ 8
#define HDQ  128
#define MR   (BQ * SQ)          // 4096 rows

// Load a 16-element bf16 fragment: 8 contiguous at p, 8 contiguous at p+blk2.
// A-fragments: blk2=8 too (16 contiguous K per lane-half, same order as B).
// B-fragments: blk2=8 (16 fully contiguous K values per lane-half).
__device__ __forceinline__ v16bf frag16(const bf16* p, int blk2) {
  v8bf lo = *reinterpret_cast<const v8bf*>(p);
  v8bf hi = *reinterpret_cast<const v8bf*>(p + blk2);
  return __builtin_shufflevector(lo, hi, 0,1,2,3,4,5,6,7,8,9,10,11,12,13,14,15);
}

__device__ __forceinline__ v8f wmma_bf16(v16bf a, v16bf b, v8f c) {
  return __builtin_amdgcn_wmma_f32_16x16x32_bf16(false, a, false, b,
                                                 (short)0, c, false, false);
}

// ---------------------------------------------------------------------------
// fp32 -> bf16 convert (coalesced both sides)
// ---------------------------------------------------------------------------
__global__ void cvt_bf16_kernel(const float* __restrict__ in,
                                bf16* __restrict__ out, size_t n) {
  size_t i  = (size_t)blockIdx.x * blockDim.x + threadIdx.x;
  size_t st = (size_t)gridDim.x * blockDim.x;
  for (; i < n; i += st) out[i] = (bf16)in[i];
}

// ---------------------------------------------------------------------------
// LDS-tiled transpose+convert: W[K][N] (f32) -> WT[N][K] (bf16).
// 32x32 tiles, 256 threads (32x8), padded LDS; coalesced reads and writes.
// ---------------------------------------------------------------------------
__global__ __launch_bounds__(256)
void transpose_cvt_kernel(const float* __restrict__ W, bf16* __restrict__ WT,
                          int Kd, int Nd) {
  __shared__ float tile[32][33];
  int ntb = Nd >> 5;
  int kt = blockIdx.x / ntb;
  int nt = blockIdx.x % ntb;
  int tx = threadIdx.x & 31;
  int ty = threadIdx.x >> 5;        // 0..7
  int k0 = kt * 32, n0 = nt * 32;
#pragma unroll
  for (int j = 0; j < 32; j += 8)
    tile[ty + j][tx] = W[(size_t)(k0 + ty + j) * Nd + n0 + tx];
  __syncthreads();
#pragma unroll
  for (int j = 0; j < 32; j += 8)
    WT[(size_t)(n0 + ty + j) * Kd + k0 + tx] = (bf16)tile[tx][ty + j];
}

// ---------------------------------------------------------------------------
// GEMM: C[M][N] (f32) = A[M][K] (bf16) * BT[N][K] (bf16, pre-transposed).
// One wave computes a 32(M) x 64(N) tile: 8 accumulators, K-steps of 32.
// All 6 fragments of a k-step are loaded before the 8 WMMAs so the loads
// issue as one clause and the waits amortize.
// ---------------------------------------------------------------------------
__global__ __launch_bounds__(256)
void gemm_bf16_kernel(const bf16* __restrict__ A, const bf16* __restrict__ BT,
                      float* __restrict__ C, int M, int N, int Kd) {
  int wave = threadIdx.x >> 5;
  int lane = threadIdx.x & 31;
  int hi = lane >> 4;      // lane half selects 16-wide K block (contiguous k = 16*hi + s)
  int ln = lane & 15;      // A: row, B: column, C: column

  int ntiles = N >> 6;
  int wid = blockIdx.x * 8 + wave;
  int tm = wid / ntiles;
  int tn = wid % ntiles;
  if (tm * 32 >= M) return;          // wave-uniform
  int m0 = tm * 32, n0 = tn * 64;

  v8f c0[4] = {}, c1[4] = {};
  const bf16* arow0 = A + (size_t)(m0 + ln) * Kd;
  const bf16* arow1 = A + (size_t)(m0 + 16 + ln) * Kd;
  const bf16* br0 = BT + (size_t)(n0 +  0 + ln) * Kd;
  const bf16* br1 = BT + (size_t)(n0 + 16 + ln) * Kd;
  const bf16* br2 = BT + (size_t)(n0 + 32 + ln) * Kd;
  const bf16* br3 = BT + (size_t)(n0 + 48 + ln) * Kd;

  for (int k = 0; k < Kd; k += 32) {
    int ka = k + hi * 16;
    int kb = k + hi * 16;
    v16bf a0 = frag16(arow0 + ka, 8);
    v16bf a1 = frag16(arow1 + ka, 8);
    v16bf b0 = frag16(br0 + kb, 8);
    v16bf b1 = frag16(br1 + kb, 8);
    v16bf b2 = frag16(br2 + kb, 8);
    v16bf b3 = frag16(br3 + kb, 8);
    c0[0] = wmma_bf16(a0, b0, c0[0]);
    c1[0] = wmma_bf16(a1, b0, c1[0]);
    c0[1] = wmma_bf16(a0, b1, c0[1]);
    c1[1] = wmma_bf16(a1, b1, c1[1]);
    c0[2] = wmma_bf16(a0, b2, c0[2]);
    c1[2] = wmma_bf16(a1, b2, c1[2]);
    c0[3] = wmma_bf16(a0, b3, c0[3]);
    c1[3] = wmma_bf16(a1, b3, c1[3]);
  }
  // C fragment: lane holds (M = r + 8*hi, N = ln) for VGPR r.
#pragma unroll
  for (int j = 0; j < 4; ++j)
#pragma unroll
    for (int r = 0; r < 8; ++r) {
      C[(size_t)(m0 + r + 8 * hi) * N + n0 + j * 16 + ln]      = c0[j][r];
      C[(size_t)(m0 + 16 + r + 8 * hi) * N + n0 + j * 16 + ln] = c1[j][r];
    }
}

// ---------------------------------------------------------------------------
// RoPE on Q (f32 in, bf16 out, same [row][NH*HD] layout).
// ---------------------------------------------------------------------------
__global__ void rope_q_kernel(const float* __restrict__ Qf, bf16* __restrict__ Qb) {
  size_t total = (size_t)MR * NHQ * 64;
  size_t i  = (size_t)blockIdx.x * blockDim.x + threadIdx.x;
  size_t st = (size_t)gridDim.x * blockDim.x;
  for (; i < total; i += st) {
    int d = (int)(i & 63);
    size_t t = i >> 6;
    int hh  = (int)(t % NHQ);
    int row = (int)(t / NHQ);
    int pos = row % SQ;
    float invf = __expf((float)d * -0.14391157f);   // 10000^(-d/64)
    float sn, cs;
    __sincosf((float)pos * invf, &sn, &cs);
    size_t base = (size_t)row * (NHQ * HDQ) + (size_t)hh * HDQ;
    float q0 = Qf[base + d];
    float q1 = Qf[base + d + 64];
    Qb[base + d]      = (bf16)(q0 * cs - q1 * sn);
    Qb[base + d + 64] = (bf16)(q1 * cs + q0 * sn);
  }
}

// RoPE on K: [row][NKV*HD] f32 -> [b][kvh][s][hd] bf16 (contiguous hd).
__global__ void rope_k_kernel(const float* __restrict__ Kf, bf16* __restrict__ Kb) {
  size_t total = (size_t)MR * NKVQ * 64;
  size_t i  = (size_t)blockIdx.x * blockDim.x + threadIdx.x;
  size_t st = (size_t)gridDim.x * blockDim.x;
  for (; i < total; i += st) {
    int d = (int)(i & 63);
    size_t t = i >> 6;
    int hh  = (int)(t % NKVQ);
    int row = (int)(t / NKVQ);
    int b = row / SQ, s = row % SQ;
    float invf = __expf((float)d * -0.14391157f);
    float sn, cs;
    __sincosf((float)s * invf, &sn, &cs);
    size_t ib = (size_t)row * (NKVQ * HDQ) + (size_t)hh * HDQ;
    size_t ob = ((size_t)(b * NKVQ + hh) * SQ + s) * HDQ;
    float k0 = Kf[ib + d];
    float k1 = Kf[ib + d + 64];
    Kb[ob + d]      = (bf16)(k0 * cs - k1 * sn);
    Kb[ob + d + 64] = (bf16)(k1 * cs + k0 * sn);
  }
}

// ---------------------------------------------------------------------------
// V: per (b,kvh) transpose [s][hd] f32 -> [hd][s] bf16 via LDS 32x32 tiles.
// ---------------------------------------------------------------------------
__global__ __launch_bounds__(256)
void cvt_v_kernel(const float* __restrict__ Vf, bf16* __restrict__ Vt) {
  __shared__ float tile[32][33];
  const int sTiles = SQ / 32;       // 64
  const int dTiles = HDQ / 32;      // 4
  int tilesPerHead = sTiles * dTiles;
  int head = blockIdx.x / tilesPerHead;       // 0..15 = b*NKV+kvh
  int t    = blockIdx.x % tilesPerHead;
  int dt = t % dTiles, st = t / dTiles;
  int b = head / NKVQ, kvh = head % NKVQ;
  const float* src = Vf + (size_t)b * SQ * (NKVQ * HDQ) + (size_t)kvh * HDQ;
  bf16* dst = Vt + (size_t)head * HDQ * SQ;
  int tx = threadIdx.x & 31;
  int ty = threadIdx.x >> 5;
  int s0 = st * 32, d0 = dt * 32;
#pragma unroll
  for (int j = 0; j < 32; j += 8)   // read: contiguous in hd
    tile[ty + j][tx] = src[(size_t)(s0 + ty + j) * (NKVQ * HDQ) + d0 + tx];
  __syncthreads();
#pragma unroll
  for (int j = 0; j < 32; j += 8)   // write: contiguous in s
    dst[(size_t)(d0 + ty + j) * SQ + s0 + tx] = (bf16)tile[tx][ty + j];
}

// ---------------------------------------------------------------------------
// Flash attention (causal, GQA). One wave = one 16-row Q tile; iterate 32-key
// tiles: 2 score WMMAs x4 K-steps -> online softmax (shfl width 16 row
// reductions) -> P staged via per-wave LDS -> 8 PV WMMAs.
// ---------------------------------------------------------------------------
__global__ __launch_bounds__(256)
void attn_kernel(const bf16* __restrict__ Q, const bf16* __restrict__ K,
                 const bf16* __restrict__ Vt, bf16* __restrict__ O) {
  __shared__ __align__(16) bf16 plds[8][16 * 32];   // per-wave P tile (16x32)

  int wave = threadIdx.x >> 5;
  int lane = threadIdx.x & 31;
  int hi = lane >> 4;
  int ln = lane & 15;

  const int qtiles = SQ / 128;                      // 16 q-chunks of 128
  int blk = blockIdx.x;
  int qc = blk % qtiles;
  int h  = (blk / qtiles) % NHQ;
  int b  = blk / (qtiles * NHQ);
  int kvh = h >> 2;                                 // NH/NKV = 4
  int qbase = qc * 128 + wave * 16;

  const bf16* Qp = Q  + (size_t)b * SQ * (NHQ * HDQ) + (size_t)h * HDQ;
  const bf16* Kp = K  + (size_t)(b * NKVQ + kvh) * SQ * HDQ;
  const bf16* Vp = Vt + (size_t)(b * NKVQ + kvh) * HDQ * SQ;

  // Preload Q A-fragments for the 4 K-steps over head_dim=128.
  v16bf qa[4];
#pragma unroll
  for (int kb = 0; kb < 4; ++kb) {
    const bf16* p = Qp + (size_t)(qbase + ln) * (NHQ * HDQ) + kb * 32 + hi * 16;
    qa[kb] = frag16(p, 8);
  }

  v8f acc[8] = {};
  float mrow[8], lrow[8];
#pragma unroll
  for (int r = 0; r < 8; ++r) { mrow[r] = -1e30f; lrow[r] = 0.0f; }
  const float sc = 0.08838834764831845f;            // 1/sqrt(128)

  int ktmax = (qbase + 15) >> 5;
  for (int kt = 0; kt <= ktmax; ++kt) {
    // S tiles: keys [kt*32, kt*32+16) and [kt*32+16, kt*32+32)
    v8f s0v = {}, s1v = {};
#pragma unroll
    for (int kb = 0; kb < 4; ++kb) {
      const bf16* bp0 = Kp + (size_t)(kt * 32 + ln)      * HDQ + kb * 32 + hi * 16;
      const bf16* bp1 = Kp + (size_t)(kt * 32 + 16 + ln) * HDQ + kb * 32 + hi * 16;
      v16bf b0 = frag16(bp0, 8);
      v16bf b1 = frag16(bp1, 8);
      s0v = wmma_bf16(qa[kb], b0, s0v);
      s1v = wmma_bf16(qa[kb], b1, s1v);
    }

    // Online softmax. Row M = r + 8*hi lives across the lane's 16-lane half.
    float alpha[8];
#pragma unroll
    for (int r = 0; r < 8; ++r) {
      int mg = qbase + r + 8 * hi;
      int n0 = kt * 32 + ln;
      float s0 = (n0      <= mg) ? s0v[r] * sc : -1e30f;
      float s1 = (n0 + 16 <= mg) ? s1v[r] * sc : -1e30f;
      float tm = fmaxf(s0, s1);
#pragma unroll
      for (int off = 1; off < 16; off <<= 1)
        tm = fmaxf(tm, __shfl_xor(tm, off, 16));
      float mn = fmaxf(mrow[r], tm);
      float a  = __expf(mrow[r] - mn);
      float p0 = __expf(s0 - mn);
      float p1 = __expf(s1 - mn);
      float ps = p0 + p1;
#pragma unroll
      for (int off = 1; off < 16; off <<= 1)
        ps += __shfl_xor(ps, off, 16);
      mrow[r] = mn;
      lrow[r] = lrow[r] * a + ps;
      alpha[r] = a;
      // Re-layout C-fragment -> LDS [row][key] for the PV A-fragment.
      plds[wave][(r + 8 * hi) * 32 + ln]      = (bf16)p0;
      plds[wave][(r + 8 * hi) * 32 + 16 + ln] = (bf16)p1;
    }
#pragma unroll
    for (int d = 0; d < 8; ++d)
#pragma unroll
      for (int r = 0; r < 8; ++r)
        acc[d][r] *= alpha[r];

    asm volatile("s_wait_dscnt 0" ::: "memory");    // P stores -> A-frag loads
    const bf16* pp = &plds[wave][ln * 32 + hi * 16];
    v16bf pa = frag16(pp, 8);
    asm volatile("" ::: "memory");                  // keep loads before next stores

#pragma unroll
    for (int d = 0; d < 8; ++d) {
      const bf16* vp = Vp + (size_t)(d * 16 + ln) * SQ + kt * 32 + hi * 16;
      v16bf vb = frag16(vp, 8);
      acc[d] = wmma_bf16(pa, vb, acc[d]);
    }
  }

  // Epilogue: O[b][s][h*128 + d] bf16.
  bf16* Op = O + (size_t)b * SQ * (NHQ * HDQ) + (size_t)h * HDQ;
#pragma unroll
  for (int d = 0; d < 8; ++d) {
#pragma unroll
    for (int r = 0; r < 8; ++r) {
      float inv = 1.0f / lrow[r];
      Op[(size_t)(qbase + r + 8 * hi) * (NHQ * HDQ) + d * 16 + ln] =
          (bf16)(acc[d][r] * inv);
    }
  }
}

// ---------------------------------------------------------------------------
// Host-side launch
// ---------------------------------------------------------------------------
extern "C" void kernel_launch(void* const* d_in, const int* in_sizes, int n_in,
                              void* d_out, int out_size, void* d_ws, size_t ws_size,
                              hipStream_t stream) {
  (void)in_sizes; (void)n_in; (void)out_size; (void)ws_size;

  const float* hidden = (const float*)d_in[0];
  /* d_in[1] = position_ids (arange, analytic in kernels) */
  const float* Wq = (const float*)d_in[2];
  const float* Wk = (const float*)d_in[3];
  const float* Wv = (const float*)d_in[4];
  const float* Wo = (const float*)d_in[5];
  float* out = (float*)d_out;

  char* ws = (char*)d_ws;
  size_t off = 0;
  auto carve = [&](size_t bytes) {
    void* p = ws + off;
    off += (bytes + 255) & ~(size_t)255;
    return p;
  };
  bf16*  hid_bf = (bf16*)carve((size_t)MR * HQ * 2);          // 33.5 MB
  bf16*  WqT    = (bf16*)carve((size_t)HQ * (NHQ*HDQ) * 2);   // 33.5 MB
  bf16*  WkT    = (bf16*)carve((size_t)HQ * (NKVQ*HDQ) * 2);  //  8.4 MB
  bf16*  WvT    = (bf16*)carve((size_t)HQ * (NKVQ*HDQ) * 2);  //  8.4 MB
  bf16*  WoT    = (bf16*)carve((size_t)(NHQ*HDQ) * HQ * 2);   // 33.5 MB
  float* Qf     = (float*)carve((size_t)MR * (NHQ*HDQ) * 4);  // 67.1 MB
  float* Kf     = (float*)carve((size_t)MR * (NKVQ*HDQ) * 4); // 16.8 MB
  float* Vf     = (float*)carve((size_t)MR * (NKVQ*HDQ) * 4); // 16.8 MB
  // Aliases over dead regions (stream order guarantees safety):
  bf16* Qb   = WqT;     // Q bf16 after RoPE (WqT dead after Q GEMM)
  bf16* Kb   = WkT;     // K bf16 after RoPE
  bf16* Vtb  = WvT;     // V^T bf16
  bf16* attn = hid_bf;  // attention output (hidden_bf dead after QKV GEMMs)

  // 1) fp32 -> bf16 convert + LDS-tiled weight transposes
  cvt_bf16_kernel<<<4096, 256, 0, stream>>>(hidden, hid_bf, (size_t)MR * HQ);
  transpose_cvt_kernel<<<(HQ/32)*((NHQ*HDQ)/32),  256, 0, stream>>>(Wq, WqT, HQ, NHQ * HDQ);
  transpose_cvt_kernel<<<(HQ/32)*((NKVQ*HDQ)/32), 256, 0, stream>>>(Wk, WkT, HQ, NKVQ * HDQ);
  transpose_cvt_kernel<<<(HQ/32)*((NKVQ*HDQ)/32), 256, 0, stream>>>(Wv, WvT, HQ, NKVQ * HDQ);
  transpose_cvt_kernel<<<((NHQ*HDQ)/32)*(HQ/32),  256, 0, stream>>>(Wo, WoT, NHQ * HDQ, HQ);

  // 2) QKV projections (WMMA bf16 -> f32); 32x64 per wave, 8 waves per block
  gemm_bf16_kernel<<<(MR/32)*((NHQ*HDQ)/64)/8,  256, 0, stream>>>(hid_bf, WqT, Qf, MR, NHQ * HDQ, HQ);
  gemm_bf16_kernel<<<(MR/32)*((NKVQ*HDQ)/64)/8, 256, 0, stream>>>(hid_bf, WkT, Kf, MR, NKVQ * HDQ, HQ);
  gemm_bf16_kernel<<<(MR/32)*((NKVQ*HDQ)/64)/8, 256, 0, stream>>>(hid_bf, WvT, Vf, MR, NKVQ * HDQ, HQ);

  // 3) RoPE + layout conversion to bf16
  rope_q_kernel<<<4096, 256, 0, stream>>>(Qf, Qb);
  rope_k_kernel<<<1024, 256, 0, stream>>>(Kf, Kb);
  cvt_v_kernel<<<(BQ*NKVQ)*(SQ/32)*(HDQ/32), 256, 0, stream>>>(Vf, Vtb);

  // 4) Causal flash attention: B*NH*(S/128) blocks, 8 waves each
  attn_kernel<<<BQ * NHQ * (SQ / 128), 256, 0, stream>>>(Qb, Kb, Vtb, attn);

  // 5) Output projection -> d_out (f32)
  gemm_bf16_kernel<<<(MR/32)*(HQ/64)/8, 256, 0, stream>>>(attn, WoT, out, MR, HQ, NHQ * HDQ);
}
